// LSTM_23459111370999
// MI455X (gfx1250) — compile-verified, matrix operand order checked by reading, not run on hardware
//
#include <hip/hip_runtime.h>
#include <hip/hip_bf16.h>

// ---------------------------------------------------------------------------
// Seq2seq LSTM (4096 batch, 50 enc + 60 dec steps, 64->512->64) for gfx1250.
//
// Each workgroup owns 32 batch rows and runs all 110 timesteps internally
// (batch rows are independent through the recurrence -> no grid sync).
// Weights are repacked once to bf16 in the WMMA B-fragment image; the
// activation panel [x|h] lives in a double-buffered bf16 LDS buffer; cell
// state c stays in f32 registers; all matmuls are v_wmma_f32_16x16x32_bf16.
//
// Round-3 change: BM 16 -> 32 with 16 waves/WG. Each wave multiplies every
// B (weight) fragment against TWO row-tile A fragments, doubling weight
// reuse (8 WMMAs per 4 weight loads) and halving per-step L2 weight traffic
// (128 WGs x 2.25 MB instead of 256 x 2.25 MB). Retained from earlier
// rounds: opaque zero-offset laundering so the loop-invariant weight stream
// is neither hoisted+spilled (round 1) nor demoted to flat loads (round 2),
// plus register double-buffering of the B fragments.
// ---------------------------------------------------------------------------

typedef __bf16 v16bf __attribute__((ext_vector_type(16)));
typedef __bf16 v8bf  __attribute__((ext_vector_type(8)));
typedef float  v8f   __attribute__((ext_vector_type(8)));

#define BATCH   4096
#define IN_DIM  64
#define HID     512
#define NG      2048              // 4*HID gate outputs
#define KTOT    576               // IN_DIM + HID
#define KSTEPS  18                // KTOT / 32
#define KL      16                // HID / 32 (decoder linear K-steps)
#define LSTRIDE 584               // KTOT + 8 pad (bf16 elems); row = 1168B (16B mult)
#define BM      32                // batch rows per workgroup
#define THREADS 512               // 16 waves

// workspace layout (bytes)
#define WP_OFF   0
#define WP_BYTES ((size_t)NG * KTOT * 2)         // 2,359,296  packed gate weights
#define WL_OFF   (WP_OFF + WP_BYTES)
#define WL_BYTES ((size_t)IN_DIM * HID * 2)      // 65,536     packed W_lin
#define BIAS_OFF (WL_OFF + WL_BYTES)
#define BIAS_BYTES ((size_t)NG * 4)              // 8,192      b_ih + b_hh

__device__ __forceinline__ float sigm_(float x) {
    return 1.0f / (1.0f + __expf(-x));
}
__device__ __forceinline__ float tanh_(float x) {
    float e = __expf(2.0f * x);
    return (e - 1.0f) / (e + 1.0f);
}

// A-fragment (16-bit 16x32, ISA 7.12.2): lane&15 = M; lane half selects
// kbase 0/8; v0..3 = K[kbase..kbase+8), v4..7 = K[kbase+16..kbase+24).
__device__ __forceinline__ v16bf load_a_frag(const __bf16* __restrict__ Ab,
                                             int m, int k /* = k0 + kbase */) {
    v8bf lo = *(const v8bf*)(Ab + m * LSTRIDE + k);
    v8bf hi = *(const v8bf*)(Ab + m * LSTRIDE + k + 16);
    return __builtin_shufflevector(lo, hi, 0, 1, 2, 3, 4, 5, 6, 7,
                                           8, 9, 10, 11, 12, 13, 14, 15);
}

// B-fragment: packed image, 32 lanes x 16 bf16 per (tile, kstep) block.
__device__ __forceinline__ v16bf load_b_frag(const __bf16* __restrict__ Wp,
                                             int tile, int ks, int nksteps, int lane) {
    return *(const v16bf*)(Wp + ((((size_t)tile) * nksteps + ks) * 32 + lane) * 16);
}

#define WMMA_BF16(a, b, c) \
    __builtin_amdgcn_wmma_f32_16x16x32_bf16(false, (a), false, (b), (short)0, (c), false, false)

// ---------------------------------------------------------------------------
// Pack [W_ih | W_hh] (f32, row = gate output n, K = [x(64) | h(512)]) into the
// bf16 B-fragment image: Wp[tile][ks][lane][e], tile = n/16 (128 tiles).
// ---------------------------------------------------------------------------
__global__ void lstm_pack_gate_w(const float* __restrict__ Wih,
                                 const float* __restrict__ Whh,
                                 __bf16* __restrict__ Wp) {
    int id = blockIdx.x * blockDim.x + threadIdx.x;
    if (id >= 128 * KSTEPS * 32) return;
    int lane  = id & 31;
    int ks    = (id >> 5) % KSTEPS;
    int tile  = id / (32 * KSTEPS);
    int kbase = (lane >> 4) * 8;
    int n     = tile * 16 + (lane & 15);
    __bf16* dst = Wp + (size_t)id * 16;
#pragma unroll
    for (int e = 0; e < 16; ++e) {
        int k = ks * 32 + kbase + (e < 8 ? e : 8 + e); // e>=8 -> kbase+16+(e-8)
        float v = (k < IN_DIM) ? Wih[(size_t)n * IN_DIM + k]
                               : Whh[(size_t)n * HID + (k - IN_DIM)];
        dst[e] = (__bf16)v;
    }
}

__global__ void lstm_pack_lin_w(const float* __restrict__ Wlin,
                                __bf16* __restrict__ Wl) {
    int id = blockIdx.x * blockDim.x + threadIdx.x;
    if (id >= 4 * KL * 32) return;
    int lane  = id & 31;
    int ks    = (id >> 5) % KL;
    int tile  = id / (32 * KL);
    int kbase = (lane >> 4) * 8;
    int n     = tile * 16 + (lane & 15);
    __bf16* dst = Wl + (size_t)id * 16;
#pragma unroll
    for (int e = 0; e < 16; ++e) {
        int k = ks * 32 + kbase + (e < 8 ? e : 8 + e);
        dst[e] = (__bf16)Wlin[(size_t)n * HID + k];
    }
}

__global__ void lstm_combine_bias(const float* __restrict__ bih,
                                  const float* __restrict__ bhh,
                                  float* __restrict__ bias) {
    int id = blockIdx.x * blockDim.x + threadIdx.x;
    if (id < NG) bias[id] = bih[id] + bhh[id];
}

// ---------------------------------------------------------------------------
// Main persistent-per-tile kernel: 512 threads = 16 waves, 32 batch rows.
// Wave w owns hidden column-groups {2w, 2w+1}; for each it updates both
// 16-row tiles, sharing every weight fragment across the two row tiles.
// Cell state: 4 C-fragments (2 groups x 2 row tiles) in f32 registers.
// ---------------------------------------------------------------------------
__global__ __launch_bounds__(THREADS) void lstm_seq2seq(
    const float* __restrict__ xseq,   // [B, S, 64]
    const __bf16* __restrict__ Wp0,   // packed gate weights
    const __bf16* __restrict__ Wl0,   // packed W_lin
    const float* __restrict__ bias,   // b_ih + b_hh [2048]
    const float* __restrict__ blin,   // [64]
    float* __restrict__ out,          // [B, P, 64]
    int S, int P) {
    __shared__ __align__(32) __bf16 A[2][BM * LSTRIDE];

    const int tid   = threadIdx.x;
    const int lane  = tid & 31;
    const int wave  = tid >> 5;          // 0..15
    const int lcol  = lane & 15;
    const int kbase = (lane >> 4) * 8;   // also the M offset for C fragments
    const int b0    = blockIdx.x * BM;

    // zero both panels (h region must start at 0; x region overwritten anyway)
    for (int i = tid; i < 2 * BM * LSTRIDE; i += THREADS)
        (&A[0][0])[i] = (__bf16)0.0f;

    // c[grp][row-tile]
    v8f c00 = {}, c01 = {}, c10 = {}, c11 = {};
    __syncthreads();

    int cur = 0;
    const int total = S + P;
    for (int step = 0; step < total; ++step) {
        // Opaque zero offset: weight addresses are not provably loop-invariant
        // (defeats LICM hoist+spill) but provenance is preserved so the loads
        // stay global_load_b128 (split loadcnt/dscnt waits, not flat).
        size_t z = 0;
        asm volatile("" : "+s"(z));
        const __bf16* __restrict__ Wp = Wp0 + z;
        const __bf16* __restrict__ Wl = Wl0 + z;

        __bf16* Ac = A[cur];
        __bf16* An = A[cur ^ 1];

        if (step < S) {
            // encoder input: x_t -> bf16 into x region of current panel
            for (int i = tid; i < BM * IN_DIM; i += THREADS) {
                int m = i >> 6, k = i & 63;
                Ac[m * LSTRIDE + k] =
                    (__bf16)xseq[((size_t)(b0 + m) * S + step) * IN_DIM + k];
            }
        } else if (wave < 8) {
            // decoder: out = h @ W_lin^T + b_lin ; wave -> (row tile, col tile)
            const int p  = step - S;
            const int rt = wave >> 2;          // 0..1
            const int ct = wave & 3;           // 0..3
            v8f acc = {};
            v16bf b = load_b_frag(Wl, ct, 0, KL, lane);
#pragma unroll
            for (int ks = 0; ks < KL; ++ks) {
                v16bf nb;
                if (ks + 1 < KL) nb = load_b_frag(Wl, ct, ks + 1, KL, lane);
                v16bf a = load_a_frag(Ac, rt * 16 + lcol, IN_DIM + ks * 32 + kbase);
                acc = WMMA_BF16(a, b, acc);
                b = nb;
            }
            const int n  = ct * 16 + lcol;
            const float bl = blin[n];
#pragma unroll
            for (int r = 0; r < 8; ++r) {
                const int m = rt * 16 + r + kbase; // C frag: M = r + 8*(lane>=16)
                float v = acc[r] + bl;
                out[((size_t)(b0 + m) * P + p) * IN_DIM + n] = v;
                Ac[m * LSTRIDE + n] = (__bf16)v;   // feed back as next input
            }
        }
        __syncthreads();

        // gates GEMM + cell update; h written to the *next* panel
#pragma unroll
        for (int grp = 0; grp < 2; ++grp) {
            const int gidx = wave * 2 + grp;       // 0..31 column-tile in each gate
            v8f ai0 = {}, af0 = {}, ag0 = {}, ao0 = {};
            v8f ai1 = {}, af1 = {}, ag1 = {}, ao1 = {};

            // software-pipelined weight stream: prologue loads ks=0
            v16bf bi = load_b_frag(Wp,  0 + gidx, 0, KSTEPS, lane);
            v16bf bf = load_b_frag(Wp, 32 + gidx, 0, KSTEPS, lane);
            v16bf bg = load_b_frag(Wp, 64 + gidx, 0, KSTEPS, lane);
            v16bf bo = load_b_frag(Wp, 96 + gidx, 0, KSTEPS, lane);
#pragma unroll
            for (int ks = 0; ks < KSTEPS; ++ks) {
                v16bf nbi, nbf, nbg, nbo;
                if (ks + 1 < KSTEPS) {
                    nbi = load_b_frag(Wp,  0 + gidx, ks + 1, KSTEPS, lane);
                    nbf = load_b_frag(Wp, 32 + gidx, ks + 1, KSTEPS, lane);
                    nbg = load_b_frag(Wp, 64 + gidx, ks + 1, KSTEPS, lane);
                    nbo = load_b_frag(Wp, 96 + gidx, ks + 1, KSTEPS, lane);
                }
                v16bf a0 = load_a_frag(Ac, lcol,      ks * 32 + kbase);
                v16bf a1 = load_a_frag(Ac, 16 + lcol, ks * 32 + kbase);
                ai0 = WMMA_BF16(a0, bi, ai0);
                ai1 = WMMA_BF16(a1, bi, ai1);
                af0 = WMMA_BF16(a0, bf, af0);
                af1 = WMMA_BF16(a1, bf, af1);
                ag0 = WMMA_BF16(a0, bg, ag0);
                ag1 = WMMA_BF16(a1, bg, ag1);
                ao0 = WMMA_BF16(a0, bo, ao0);
                ao1 = WMMA_BF16(a1, bo, ao1);
                bi = nbi; bf = nbf; bg = nbg; bo = nbo;
            }
            const int n = gidx * 16 + lcol;
            const float Bi = bias[n];
            const float Bf = bias[HID + n];
            const float Bg = bias[2 * HID + n];
            const float Bo = bias[3 * HID + n];

#pragma unroll
            for (int rt = 0; rt < 2; ++rt) {
                v8f cc = (grp == 0) ? (rt == 0 ? c00 : c01)
                                    : (rt == 0 ? c10 : c11);
                const v8f& xi = rt == 0 ? ai0 : ai1;
                const v8f& xf = rt == 0 ? af0 : af1;
                const v8f& xg = rt == 0 ? ag0 : ag1;
                const v8f& xo = rt == 0 ? ao0 : ao1;
#pragma unroll
                for (int r = 0; r < 8; ++r) {
                    float iv = sigm_(xi[r] + Bi);
                    float fv = sigm_(xf[r] + Bf);
                    float gv = tanh_(xg[r] + Bg);
                    float ov = sigm_(xo[r] + Bo);
                    float cn = fv * cc[r] + iv * gv;
                    cc[r] = cn;
                    float hv = ov * tanh_(cn);
                    An[(rt * 16 + r + kbase) * LSTRIDE + IN_DIM + n] = (__bf16)hv;
                }
                if (grp == 0) { if (rt == 0) c00 = cc; else c01 = cc; }
                else          { if (rt == 0) c10 = cc; else c11 = cc; }
            }
        }
        __syncthreads();
        cur ^= 1;
    }
}

// ---------------------------------------------------------------------------
extern "C" void kernel_launch(void* const* d_in, const int* in_sizes, int n_in,
                              void* d_out, int out_size, void* d_ws, size_t ws_size,
                              hipStream_t stream) {
    (void)n_in; (void)ws_size;
    const float* xseq = (const float*)d_in[0];   // input_seq [B,S,64]
    const float* Wih  = (const float*)d_in[1];   // [2048,64]
    const float* Whh  = (const float*)d_in[2];   // [2048,512]
    const float* bih  = (const float*)d_in[3];   // [2048]
    const float* bhh  = (const float*)d_in[4];   // [2048]
    const float* Wlin = (const float*)d_in[5];   // [64,512]
    const float* blin = (const float*)d_in[6];   // [64]
    float* out = (float*)d_out;

    char* ws = (char*)d_ws;
    __bf16* Wp   = (__bf16*)(ws + WP_OFF);
    __bf16* Wl   = (__bf16*)(ws + WL_OFF);
    float*  bias = (float*)(ws + BIAS_OFF);

    const int S = in_sizes[0] / (BATCH * IN_DIM);   // 50
    const int P = out_size    / (BATCH * IN_DIM);   // 60

    lstm_pack_gate_w<<<(128 * KSTEPS * 32 + 255) / 256, 256, 0, stream>>>(Wih, Whh, Wp);
    lstm_pack_lin_w<<<(4 * KL * 32 + 255) / 256, 256, 0, stream>>>(Wlin, Wl);
    lstm_combine_bias<<<(NG + 255) / 256, 256, 0, stream>>>(bih, bhh, bias);

    lstm_seq2seq<<<BATCH / BM, THREADS, 0, stream>>>(xseq, Wp, Wl, bias, blin, out, S, P);
}
